// MultiHeadAttention_29807073034465
// MI455X (gfx1250) — matrix operand order repaired and checked by reading, not compiled, not run on hardware
//
#include <hip/hip_runtime.h>
#include <hip/hip_bf16.h>

// ---------------------------------------------------------------------------
// MultiHeadAttention for MI455X (gfx1250), wave32 + WMMA bf16 (split hi/lo
// for ~f32 precision). QK^T GEMM fused with masked softmax: scores never
// touch HBM (16KB tile in LDS); p_attn (537 MB) streamed with NT stores.
// Shapes: N=8, T=256, H=8, KQ=128, V=32.
// ---------------------------------------------------------------------------

typedef __attribute__((ext_vector_type(16))) __bf16 v16bf;
typedef __attribute__((ext_vector_type(8)))  float  v8f;

#define WMMA_BF16(A, B, C) \
  __builtin_amdgcn_wmma_f32_16x16x32_bf16(false, (A), false, (B), (short)0, (C), false, false)

__device__ __forceinline__ void load_a_f32_split(const float* __restrict__ row,
                                                 int kb, int hi, v16bf& ah, v16bf& al) {
  // A operand 16x32 bf16: lane holds row r; lanes<16 -> K 0..15,
  // lanes>=16 -> K 16..31 (relative to kb).
#pragma unroll
  for (int j = 0; j < 16; ++j) {
    float x0 = row[kb + 16 * hi + j];
    __bf16 h0 = (__bf16)x0;
    ah[j] = h0; al[j] = (__bf16)(x0 - (float)h0);
  }
}

__device__ __forceinline__ void load_b_f32_split(const float* __restrict__ row,
                                                 int kb, int hi, v16bf& bh, v16bf& bl) {
  // B operand 32x16 bf16: lane holds column r (== row r of rhs when doing A*B^T);
  // lanes<16 -> K 0..15, lanes>=16 -> K 16..31 (relative to kb).
#pragma unroll
  for (int j = 0; j < 16; ++j) {
    float x = row[kb + 16 * hi + j];
    __bf16 h = (__bf16)x;
    bh[j] = h; bl[j] = (__bf16)(x - (float)h);
  }
}

__device__ __forceinline__ v16bf load_a_bf(const __bf16* __restrict__ row, int kb, int hi) {
  v16bf a;
#pragma unroll
  for (int j = 0; j < 16; ++j) a[j] = row[kb + 16 * hi + j];
  return a;
}

__device__ __forceinline__ v16bf load_b_bf(const __bf16* __restrict__ row, int kb, int hi) {
  v16bf b;
#pragma unroll
  for (int j = 0; j < 16; ++j) b[j] = row[kb + 16 * hi + j];
  return b;
}

// ---------------------------------------------------------------------------
// Kernel 1: Q/K projections.  C(2048x1024) = X(2048x128) * W^T(128x1024) + b.
// One wave per 16x16 tile; K=128 -> 4 k-steps x 3 split-WMMAs.
// Output stored as bf16 hi/lo pairs in [n][h][t][d] layout (K-contiguous).
// ---------------------------------------------------------------------------
__global__ void __launch_bounds__(256)
k_proj(const float* __restrict__ query, const float* __restrict__ key,
       const float* __restrict__ Wq, const float* __restrict__ bq,
       const float* __restrict__ Wk, const float* __restrict__ bk,
       __bf16* __restrict__ hq_hi, __bf16* __restrict__ hq_lo,
       __bf16* __restrict__ hk_hi, __bf16* __restrict__ hk_lo) {
  int wave = blockIdx.x * 8 + (threadIdx.x >> 5);
  int lane = threadIdx.x & 31;
  int mat = wave >> 13;          // 0 = query, 1 = key
  int rem = wave & 8191;
  int mt = rem >> 6;             // 128 M-tiles over N*T=2048
  int nt = rem & 63;             // 64 N-tiles over H*KQ=1024
  const float* X  = mat ? key : query;
  const float* W  = mat ? Wk : Wq;
  const float* Bv = mat ? bk : bq;
  __bf16* Dh = mat ? hk_hi : hq_hi;
  __bf16* Dl = mat ? hk_lo : hq_lo;

  int r = lane & 15, hi = lane >> 4;
  const float* arow = X + (size_t)(mt * 16 + r) * 128;
  const float* wrow = W + (size_t)(nt * 16 + r) * 128;

  v8f acc = {};
#pragma unroll
  for (int kb = 0; kb < 128; kb += 32) {
    v16bf ah, al, bh, bl;
    load_a_f32_split(arow, kb, hi, ah, al);
    load_b_f32_split(wrow, kb, hi, bh, bl);
    acc = WMMA_BF16(ah, bh, acc);
    acc = WMMA_BF16(ah, bl, acc);
    acc = WMMA_BF16(al, bh, acc);
  }
  int   col  = nt * 16 + r;          // column of C == per-lane output column
  float bias = Bv[col];
  int h = col >> 7, d = col & 127;
#pragma unroll
  for (int i = 0; i < 8; ++i) {
    int m = mt * 16 + 8 * hi + i;    // C row M = i + 8*hi
    int n = m >> 8, t = m & 255;
    float c = acc[i] + bias;
    __bf16 chi = (__bf16)c;
    __bf16 clo = (__bf16)(c - (float)chi);
    size_t idx = (((size_t)(n * 8 + h) * 256) + t) * 128 + d;
    Dh[idx] = chi; Dl[idx] = clo;
  }
}

// ---------------------------------------------------------------------------
// Kernel 2 (fused): per (n,h, 16-q tile):
//   WMMA QK^T tile (16x256) -> LDS, max/exp in place, masked per-(q,v)
//   softmax denominators + fused out = sum_k p*value, NT-store p_attn.
// Grid: 64 (n,h) x 16 q-tiles = 1024 blocks of 256 threads (8 waves).
// ---------------------------------------------------------------------------
__global__ void __launch_bounds__(256)
k_attn(const __bf16* __restrict__ hq_hi, const __bf16* __restrict__ hq_lo,
       const __bf16* __restrict__ hk_hi, const __bf16* __restrict__ hk_lo,
       const int* __restrict__ mask, const float* __restrict__ value,
       float* __restrict__ p_attn, float* __restrict__ outbuf) {
  const float SCALE = 0.088388347648318447f;  // 1/sqrt(128)
  __shared__ float esh[16][256];     // scores, then exp(s - max), 16 KB
  __shared__ float pmax[16][16];
  __shared__ float dsh[16][8][32];   // denominator partials, 16 KB
  __shared__ float osh[16][8][32];   // out partials, 16 KB
  __shared__ float invsh[16][32];

  int blk = blockIdx.x;              // blk = nh*16 + qt
  int nh = blk >> 4;
  int qt = blk & 15;
  int n = nh >> 3, h = nh & 7;
  int t = threadIdx.x;
  int wid = t >> 5, lane = t & 31;
  int r = lane & 15, hi = lane >> 4;

  // ---- Phase A: S[0:16][0:256] = (hq_tile * hk^T) * SCALE via WMMA ----
  size_t qoff = ((size_t)nh * 256 + qt * 16 + r) * 128;
  const __bf16* qhp = hq_hi + qoff;
  const __bf16* qlp = hq_lo + qoff;
  v16bf Ah[4], Al[4];
#pragma unroll
  for (int s4 = 0; s4 < 4; ++s4) {   // A operands loaded once, reused both k-tiles
    Ah[s4] = load_a_bf(qhp, s4 * 32, hi);
    Al[s4] = load_a_bf(qlp, s4 * 32, hi);
  }
#pragma unroll
  for (int half = 0; half < 2; ++half) {
    int kt = wid + half * 8;         // this wave's 16x16 k-tile
    size_t koff = ((size_t)nh * 256 + kt * 16 + r) * 128;
    const __bf16* khp = hk_hi + koff;
    const __bf16* klp = hk_lo + koff;
    v8f acc = {};
#pragma unroll
    for (int s4 = 0; s4 < 4; ++s4) {
      v16bf Bh = load_b_bf(khp, s4 * 32, hi);
      v16bf Bl = load_b_bf(klp, s4 * 32, hi);
      acc = WMMA_BF16(Ah[s4], Bh, acc);
      acc = WMMA_BF16(Ah[s4], Bl, acc);
      acc = WMMA_BF16(Al[s4], Bh, acc);
    }
#pragma unroll
    for (int i = 0; i < 8; ++i)
      esh[8 * hi + i][kt * 16 + r] = acc[i] * SCALE;
  }
  __syncthreads();

  // ---- Phase B1: per-row max, then exp in place ----
  {
    int q = t >> 4, kk = t & 15;     // 16 threads per q-row
    float mx = -3.0e38f;
#pragma unroll
    for (int j = 0; j < 16; ++j) mx = fmaxf(mx, esh[q][kk + 16 * j]);
    pmax[q][kk] = mx;
    __syncthreads();
#pragma unroll
    for (int j = 0; j < 16; ++j) mx = fmaxf(mx, pmax[q][j]);
#pragma unroll
    for (int j = 0; j < 16; ++j) {
      int k = kk + 16 * j;
      esh[q][k] = __expf(esh[q][k] - mx);
    }
  }
  __syncthreads();

  // ---- Phase B2: thread (rr,v) owns 32 k's; accumulate den/op for all 16 q.
  //      mask/value read exactly once per thread; mask cached as bitmask. ----
  int v = lane, rr = wid;
  const int*   mrow = mask  + (size_t)n * 8192;   // [k][v]
  const float* vrow = value + (size_t)n * 8192;   // [k][v]
  unsigned mb = 0;
  float den[16], op[16];
#pragma unroll
  for (int qq = 0; qq < 16; ++qq) { den[qq] = 0.f; op[qq] = 0.f; }
  for (int j = 0; j < 32; ++j) {
    int k = rr * 32 + j;
    int mm = mrow[(size_t)k * 32 + v];            // coalesced across lanes
    if (mm) {
      mb |= (1u << j);
      float val = vrow[(size_t)k * 32 + v];
#pragma unroll
      for (int qq = 0; qq < 16; ++qq) {           // esh broadcast read
        float e = esh[qq][k];
        den[qq] += e;
        op[qq]  += e * val;
      }
    }
  }
#pragma unroll
  for (int qq = 0; qq < 16; ++qq) { dsh[qq][rr][v] = den[qq]; osh[qq][rr][v] = op[qq]; }
  __syncthreads();

  // ---- Phase B3: reduce 8 partials for 512 (q,v) pairs; emit out & 1/den ----
#pragma unroll
  for (int i = 0; i < 2; ++i) {
    int p = t + 256 * i;
    int qq = p >> 5, vv = p & 31;
    float dt = 0.f, ot = 0.f;
#pragma unroll
    for (int r2 = 0; r2 < 8; ++r2) { dt += dsh[qq][r2][vv]; ot += osh[qq][r2][vv]; }
    float inv = dt > 0.f ? 1.f / dt : 0.f;
    invsh[qq][vv] = inv;
    outbuf[(size_t)(n * 256 + qt * 16 + qq) * 256 + h * 32 + vv] = ot * inv;
  }
  __syncthreads();

  // ---- Phase B4: stream p_attn (512 KB per block) with NT stores ----
  float* pbase = p_attn + ((size_t)nh * 256 + qt * 16) * 8192;  // [q][k][v]
  for (int qq = 0; qq < 16; ++qq) {
    float inv = invsh[qq][v];
    float* prow = pbase + (size_t)qq * 8192;
#pragma unroll
    for (int j = 0; j < 32; ++j) {
      int k = rr * 32 + j;
      float pv = ((mb >> j) & 1u) ? esh[qq][k] * inv : 0.f;
      __builtin_nontemporal_store(pv, prow + (size_t)k * 32 + v);  // 128B/row coalesced
    }
  }
}

// ---------------------------------------------------------------------------
// Kernel 3: out_rep(2048x32) = out(2048x256) * Wo^T(256x32) + bo.
// 256 wave-tiles, K=256 -> 8 k-steps x 3 split-WMMAs.
// ---------------------------------------------------------------------------
__global__ void __launch_bounds__(256)
k_oproj(const float* __restrict__ outbuf, const float* __restrict__ Wo,
        const float* __restrict__ bo, float* __restrict__ out_rep) {
  int wave = blockIdx.x * 8 + (threadIdx.x >> 5);
  int lane = threadIdx.x & 31;
  int mt = wave >> 1;                 // 128 M-tiles
  int nt = wave & 1;                  // 2 N-tiles over V=32
  int r = lane & 15, hi = lane >> 4;
  const float* arow = outbuf + (size_t)(mt * 16 + r) * 256;
  const float* wrow = Wo     + (size_t)(nt * 16 + r) * 256;

  v8f acc = {};
#pragma unroll
  for (int kb = 0; kb < 256; kb += 32) {
    v16bf ah, al, bh, bl;
    load_a_f32_split(arow, kb, hi, ah, al);
    load_b_f32_split(wrow, kb, hi, bh, bl);
    acc = WMMA_BF16(ah, bh, acc);
    acc = WMMA_BF16(ah, bl, acc);
    acc = WMMA_BF16(al, bh, acc);
  }
  float bias = bo[nt * 16 + r];
#pragma unroll
  for (int i = 0; i < 8; ++i) {
    int m = mt * 16 + 8 * hi + i;
    out_rep[(size_t)m * 32 + nt * 16 + r] = acc[i] + bias;
  }
}

// ---------------------------------------------------------------------------
extern "C" void kernel_launch(void* const* d_in, const int* in_sizes, int n_in,
                              void* d_out, int out_size, void* d_ws, size_t ws_size,
                              hipStream_t stream) {
  (void)in_sizes; (void)n_in; (void)out_size; (void)ws_size;
  const float* value = (const float*)d_in[0];
  const float* key   = (const float*)d_in[1];
  const float* query = (const float*)d_in[2];
  const int*   mask  = (const int*)  d_in[3];
  const float* Wq    = (const float*)d_in[4];
  const float* bq    = (const float*)d_in[5];
  const float* Wk    = (const float*)d_in[6];
  const float* bk    = (const float*)d_in[7];
  const float* Wo    = (const float*)d_in[8];
  const float* bo    = (const float*)d_in[9];

  float* out_rep = (float*)d_out;                    // 8*256*32 = 65536 floats
  float* p_attn  = out_rep + (size_t)65536;          // 8*8*256*256*32 floats

  // Workspace (18 MB): bf16 hi/lo projections + fused attention output.
  const size_t PROJ = 2097152;                       // N*H*T*KQ elements
  char* ws = (char*)d_ws;
  __bf16* hq_hi = (__bf16*)ws;
  __bf16* hq_lo = hq_hi + PROJ;
  __bf16* hk_hi = hq_lo + PROJ;
  __bf16* hk_lo = hk_hi + PROJ;
  float*  outbuf = (float*)(ws + 4 * PROJ * sizeof(__bf16));   // 2 MB

  k_proj <<<2048, 256, 0, stream>>>(query, key, Wq, bq, Wk, bk,
                                    hq_hi, hq_lo, hk_hi, hk_lo);
  k_attn <<<1024, 256, 0, stream>>>(hq_hi, hq_lo, hk_hi, hk_lo,
                                    mask, value, p_attn, outbuf);
  k_oproj<<<32,   256, 0, stream>>>(outbuf, Wo, bo, out_rep);
}
